// PositionalPhasorStream_37082747633851
// MI455X (gfx1250) — matrix operand order repaired and checked by reading, not compiled, not run on hardware
//
#include <hip/hip_runtime.h>
#include <hip/hip_bf16.h>

// ---------------------------------------------------------------------------
// PositionalPhasorStream for MI455X (gfx1250, wave32, WMMA)
//   value = x @ W^T + b            -> bf16 WMMA GEMM, fp32 accumulate,
//                                     W tiles staged in LDS via async copy
//   mem   = cumsum(value * e^{ip}) -> segmented parallel scan (16 segs x 256)
//   out   = Re(mem * e^{-ip}) / sqrt(pos)
// ---------------------------------------------------------------------------

#define DIM     1024
#define LSEQ    4096
#define BATCH   4
#define NROWS   (BATCH * LSEQ)      // 16384 GEMM rows
#define NSEG    16
#define SEGLEN  (LSEQ / NSEG)       // 256

typedef __bf16 bf16;
typedef __attribute__((ext_vector_type(8)))  bf16  v8bf;
typedef __attribute__((ext_vector_type(16))) bf16  v16bf;
typedef __attribute__((ext_vector_type(8)))  float v8f;
typedef __attribute__((ext_vector_type(4)))  float v4f;
typedef int v4i __attribute__((vector_size(16)));   // gcc-style int4 (builtin sig)

#define GAS __attribute__((address_space(1)))
#define LAS __attribute__((address_space(3)))

#if __has_builtin(__builtin_amdgcn_global_load_async_to_lds_b128)
#define HAS_ASYNC_LDS 1
#else
#define HAS_ASYNC_LDS 0
#endif

#if __has_builtin(__builtin_amdgcn_s_wait_asynccnt)
#define WAIT_ASYNC(n) __builtin_amdgcn_s_wait_asynccnt(n)
#else
#define WAIT_ASYNC(n) asm volatile("s_wait_asynccnt %0" ::"i"(n))
#endif

// ---------------------------------------------------------------------------
// K0: fp32 -> bf16 bulk convert (vectorized float4 loads)
// ---------------------------------------------------------------------------
__global__ __launch_bounds__(256)
void k_convert_bf16(const float* __restrict__ in, bf16* __restrict__ out, int n4)
{
    int i = blockIdx.x * blockDim.x + threadIdx.x;
    int stride = gridDim.x * blockDim.x;
    const v4f* in4 = (const v4f*)in;
    for (; i < n4; i += stride) {
        v4f v = in4[i];
        int base = i * 4;
        out[base + 0] = (bf16)v.x;
        out[base + 1] = (bf16)v.y;
        out[base + 2] = (bf16)v.z;
        out[base + 3] = (bf16)v.w;
    }
}

// ---------------------------------------------------------------------------
// K1: GEMM value[r][e] = sum_k xb[r][k] * wb[e][k] + bias[e]
//     WG = 8 waves, tile 128(M) x 128(N); wave tile 16 x 128 (8 acc).
//     W tile (128 rows x 32 k, 8 KB) double-buffered in LDS via async copy.
//     LDS rows: 32 bf16 (64B) with XOR swizzle of the 16B sub-chunk by row
//     bits [3:2] -> conflict-free ds_load_b128 B-fragment reads.
// ---------------------------------------------------------------------------
__global__ __launch_bounds__(256)
void k_gemm_wmma(const bf16* __restrict__ xb, const bf16* __restrict__ wb,
                 const float* __restrict__ bias, float* __restrict__ value)
{
    __shared__ bf16 Bt[2][128 * 32];               // 2 x 8 KB

    const int tid     = threadIdx.x;
    const int lane    = tid & 31;
    const int wave    = tid >> 5;
    const int l15     = lane & 15;
    const int halfsel = lane >> 4;                 // 0: lanes 0-15, 1: 16-31

    const int m0 = blockIdx.x * 128 + wave * 16;   // 16-row strip per wave
    const int n0 = blockIdx.y * 128;               // 128-col strip per WG

    const bf16* __restrict__ arow = xb + (size_t)(m0 + l15) * DIM;
    const int akoff = halfsel * 8;                 // A per-lane k sub-offset

    v8f acc[8] = {};                               // 16x128 fp32 accumulators

    // ---- async copy of one 128x32 W tile into LDS buffer `buf` ----
    auto copy_tile = [&](int buf, int k0) {
        #pragma unroll
        for (int c = 0; c < 2; ++c) {              // 512 16B chunks / 256 thr
            const int idx  = c * 256 + tid;
            const int r    = idx >> 2;             // row 0..127
            const int kof  = (idx & 3) * 8;        // {0,8,16,24} elements
            const int sw   = ((r >> 2) & 3) * 8;   // XOR swizzle
            const bf16* gp = wb + (size_t)(n0 + r) * DIM + k0 + kof;
            bf16*       lp = &Bt[buf][r * 32 + (kof ^ sw)];
#if HAS_ASYNC_LDS
            __builtin_amdgcn_global_load_async_to_lds_b128(
                (GAS v4i*)gp, (LAS v4i*)lp, 0, 0);
#else
            *(v8bf*)lp = *(const v8bf*)gp;         // register-staged fallback
#endif
        }
    };

    copy_tile(0, 0);

    // software-pipelined A fragment
    v8bf alo = *(const v8bf*)(arow + akoff);
    v8bf ahi = *(const v8bf*)(arow + 16 + akoff);

    int buf = 0;
    for (int k0 = 0; k0 < DIM; k0 += 32) {
        const bool haveNext = (k0 + 32) < DIM;
        if (haveNext) {
            copy_tile(buf ^ 1, k0 + 32);           // stream next W tile
            __builtin_prefetch((const void*)(arow + k0 + 32 + akoff), 0, 1);
        }
#if HAS_ASYNC_LDS
        if (haveNext) WAIT_ASYNC(2);               // current tile landed
        else          WAIT_ASYNC(0);
#endif
        __syncthreads();                           // LDS visible to all waves

        v16bf afrag;
        #pragma unroll
        for (int i = 0; i < 8; ++i) { afrag[i] = alo[i]; afrag[8 + i] = ahi[i]; }

        // preload next A fragment (hidden behind WMMAs)
        v8bf aloN = alo, ahiN = ahi;
        if (haveNext) {
            aloN = *(const v8bf*)(arow + k0 + 32 + akoff);
            ahiN = *(const v8bf*)(arow + k0 + 48 + akoff);
        }

        #pragma unroll
        for (int ns = 0; ns < 8; ++ns) {
            const int r  = ns * 16 + l15;
            const int sw = ((r >> 2) & 3) * 8;
            const bf16* base = &Bt[buf][r * 32];
            v8bf blo = *(const v8bf*)(base + ((halfsel * 8 + 0) ^ sw));
            v8bf bhi = *(const v8bf*)(base + ((halfsel * 8 + 16) ^ sw));
            v16bf bfrag;
            #pragma unroll
            for (int i = 0; i < 8; ++i) { bfrag[i] = blo[i]; bfrag[8 + i] = bhi[i]; }

            acc[ns] = __builtin_amdgcn_wmma_f32_16x16x32_bf16(
                false, afrag, false, bfrag, (short)0, acc[ns], false, false);
        }

        __syncthreads();                           // reads done before rewrite
        alo = aloN; ahi = ahiN;
        buf ^= 1;
    }

    // Epilogue: add bias, store fp32.
    // C/D layout: VGPR r -> row m0 + r + 8*halfsel, col = n-tile + l15.
    #pragma unroll
    for (int ns = 0; ns < 8; ++ns) {
        const int col = n0 + ns * 16 + l15;
        const float bv = bias[col];
        float* outp = value + (size_t)(m0 + halfsel * 8) * DIM + col;
        #pragma unroll
        for (int r = 0; r < 8; ++r)
            outp[(size_t)r * DIM] = acc[ns][r] + bv;
    }
}

// ---------------------------------------------------------------------------
// K2: per-segment sums of value*cos(p) and value*sin(p).
// ---------------------------------------------------------------------------
__global__ __launch_bounds__(256)
void k_seg_reduce(const float* __restrict__ value, const float* __restrict__ phases,
                  float* __restrict__ segr, float* __restrict__ segi)
{
    const int epart = blockIdx.x & 3;
    const int s     = (blockIdx.x >> 2) & 15;
    const int batch = blockIdx.x >> 6;
    const int e     = epart * 256 + threadIdx.x;

    const float* vp = value  + ((size_t)batch * LSEQ + s * SEGLEN) * DIM + e;
    const float* pp = phases + (size_t)(s * SEGLEN) * DIM + e;

    float sr = 0.f, si = 0.f;
    for (int l = 0; l < SEGLEN; ++l) {
        float v  = vp[(size_t)l * DIM];
        float ph = pp[(size_t)l * DIM];
        float c, sn;
        __sincosf(ph, &sn, &c);
        sr += v * c;
        si += v * sn;
    }
    const size_t idx = ((size_t)batch * NSEG + s) * DIM + e;
    segr[idx] = sr;
    segi[idx] = si;
}

// ---------------------------------------------------------------------------
// K3: exclusive scan over the 16 segments for every (batch, e) column.
// ---------------------------------------------------------------------------
__global__ __launch_bounds__(256)
void k_seg_scan(float* __restrict__ segr, float* __restrict__ segi)
{
    const int col   = blockIdx.x * blockDim.x + threadIdx.x;   // BATCH*DIM cols
    const int batch = col >> 10;
    const int e     = col & (DIM - 1);
    float rr = 0.f, ri = 0.f;
    for (int s = 0; s < NSEG; ++s) {
        const size_t idx = ((size_t)batch * NSEG + s) * DIM + e;
        const float tr = segr[idx], ti = segi[idx];
        segr[idx] = rr;                 // exclusive prefix
        segi[idx] = ri;
        rr += tr;
        ri += ti;
    }
}

// ---------------------------------------------------------------------------
// K4: final streaming pass -- in-segment running prefix + carry, retrieve,
//     normalize by 1/sqrt(position).
// ---------------------------------------------------------------------------
__global__ __launch_bounds__(256)
void k_scan_final(const float* __restrict__ value, const float* __restrict__ phases,
                  const float* __restrict__ segr, const float* __restrict__ segi,
                  float* __restrict__ out)
{
    const int epart = blockIdx.x & 3;
    const int s     = (blockIdx.x >> 2) & 15;
    const int batch = blockIdx.x >> 6;
    const int e     = epart * 256 + threadIdx.x;

    const size_t base = ((size_t)batch * LSEQ + s * SEGLEN) * DIM + e;
    const float* vp = value + base;
    const float* pp = phases + (size_t)(s * SEGLEN) * DIM + e;
    float*       op = out + base;

    const size_t idx = ((size_t)batch * NSEG + s) * DIM + e;
    float rr = segr[idx];               // carry from preceding segments
    float ri = segi[idx];

    for (int l = 0; l < SEGLEN; ++l) {
        float v  = vp[(size_t)l * DIM];
        float ph = pp[(size_t)l * DIM];
        float c, sn;
        __sincosf(ph, &sn, &c);
        rr += v * c;
        ri += v * sn;
        const float pos = (float)(s * SEGLEN + l + 1);
        op[(size_t)l * DIM] = (rr * c + ri * sn) * __frsqrt_rn(pos);
    }
}

// ---------------------------------------------------------------------------
// Launcher
// ---------------------------------------------------------------------------
extern "C" void kernel_launch(void* const* d_in, const int* in_sizes, int n_in,
                              void* d_out, int out_size, void* d_ws, size_t ws_size,
                              hipStream_t stream)
{
    const float* x      = (const float*)d_in[0];   // [4, 4096, 1024]
    const float* phases = (const float*)d_in[1];   // [8192, 1024] (first 4096 used)
    const float* W      = (const float*)d_in[2];   // [1024, 1024]
    const float* bias   = (const float*)d_in[3];   // [1024]
    float*       out    = (float*)d_out;           // [4, 4096, 1024]

    // Workspace layout (bytes)
    char* ws = (char*)d_ws;
    bf16*  xb    = (bf16*)(ws);                                   // 32 MiB
    bf16*  wb    = (bf16*)(ws + (size_t)NROWS * DIM * 2);         //  2 MiB
    float* value = (float*)(ws + (size_t)NROWS * DIM * 2
                               + (size_t)DIM * DIM * 2);          // 64 MiB
    float* segr  = (float*)((char*)value + (size_t)NROWS * DIM * 4);
    float* segi  = segr + (size_t)BATCH * NSEG * DIM;

    // K0: convert x and W to bf16
    k_convert_bf16<<<2048, 256, 0, stream>>>(x, xb, NROWS * DIM / 4);
    k_convert_bf16<<<512,  256, 0, stream>>>(W, wb, DIM * DIM / 4);

    // K1: WMMA GEMM (grid: 16384/128 x 1024/128)
    dim3 ggrid(NROWS / 128, DIM / 128);
    k_gemm_wmma<<<ggrid, 256, 0, stream>>>(xb, wb, bias, value);

    // K2: segment sums   (4 batches * 16 segs * 4 e-parts = 256 blocks)
    k_seg_reduce<<<BATCH * NSEG * 4, 256, 0, stream>>>(value, phases, segr, segi);

    // K3: exclusive scan over segments (4096 columns)
    k_seg_scan<<<(BATCH * DIM) / 256, 256, 0, stream>>>(segr, segi);

    // K4: final scan + retrieval + normalization
    k_scan_final<<<BATCH * NSEG * 4, 256, 0, stream>>>(value, phases, segr, segi, out);
}
